// M3GNetGraphConv_42056319762561
// MI455X (gfx1250) — matrix operand order repaired and checked by reading, not compiled, not run on hardware
//
#include <hip/hip_runtime.h>
#include <hip/hip_bf16.h>

typedef float  v8f   __attribute__((ext_vector_type(8)));
typedef __bf16 v16bf __attribute__((ext_vector_type(16)));
typedef __bf16 bf4   __attribute__((ext_vector_type(4)));

#define DEV __device__ __forceinline__

constexpr int       NN   = 50000;
constexpr long long E    = 1600000;
constexpr int       D    = 64;
constexpr int       DIN  = 192;     // 3*D
constexpr int       NRBF = 9;
constexpr int       WAVES = 8;
constexpr int       EPB   = WAVES * 16;   // 128 edges per block

// ---------------- LDS layout (bytes) ----------------
constexpr int OFF_WEL1 = 0;                     // 64x192 bf16
constexpr int OFF_WEG1 = OFF_WEL1 + 12288 * 2;  // 24576
constexpr int OFF_WEL2 = OFF_WEG1 + 12288 * 2;  // 49152  (64x64)
constexpr int OFF_WEG2 = OFF_WEL2 + 4096 * 2;   // 57344
constexpr int OFF_WNL1 = OFF_WEG2 + 4096 * 2;   // 65536
constexpr int OFF_WNG1 = OFF_WNL1 + 12288 * 2;  // 90112
constexpr int OFF_WNL2 = OFF_WNG1 + 12288 * 2;  // 114688
constexpr int OFF_WNG2 = OFF_WNL2 + 4096 * 2;   // 122880
constexpr int OFF_BIAS = OFF_WNG2 + 4096 * 2;   // 131072 : 8*64 f32
constexpr int OFF_RBFW = OFF_BIAS + 8 * 64 * 4; // 133120 : 2*64*12 f32
constexpr int OFF_PW   = OFF_RBFW + 2 * 64 * 12 * 4; // 139264
// per-wave region
constexpr int PW_X    = 0;        // 16*192 bf16 = 6144
constexpr int PW_H    = 6144;     // 16*64  bf16 = 2048
constexpr int PW_G    = 8192;     // 2048
constexpr int PW_RBF  = 10240;    // 16*12 f32 = 768
constexpr int PW_DST  = 11008;    // 16 int = 64
constexpr int PW_STRIDE = 11136;
constexpr int SMEM_BYTES = OFF_PW + WAVES * PW_STRIDE;  // 228352 (< 320KB/WGP)

// bias slot order in LDS
enum { B_EL1 = 0, B_EG1, B_EL2, B_EG2, B_NL1, B_NG1, B_NL2, B_NG2 };

DEV float silu_f(float x) { return x / (1.0f + __expf(-x)); }
DEV float sigm_f(float x) { return 1.0f / (1.0f + __expf(-x)); }

// A fragment (16xK tile, K-block kb of 32) per ISA 7.12.2 16-bit A layout:
// lanes 0-15: row=lane, K = kb*32 + {0..7, 16..23}; lanes 16-31: row=lane-16, +8
DEV v16bf lds_frag_a(const __bf16* base, int ldk, int kb) {
  int lane = threadIdx.x & 31;
  int row  = lane & 15;
  int off  = (lane < 16) ? 0 : 8;
  const __bf16* p = base + row * ldk + kb * 32 + off;
  union { v16bf v; uint4 q[2]; } u;
  u.q[0] = *(const uint4*)(p);
  u.q[1] = *(const uint4*)(p + 16);
  return u.v;
}

// B fragment (32x16, K-block kb, N-tile n) from row-major W[d][k] bf16:
// lane = output column; lanes 0-15 hold K {0..7, 16..23}, lanes 16-31 hold K {8..15, 24..31}
DEV v16bf lds_frag_b(const __bf16* wbase, int ldk, int n, int kb) {
  int lane = threadIdx.x & 31;
  int col  = lane & 15;
  int off  = (lane < 16) ? 0 : 8;
  const __bf16* p = wbase + (n * 16 + col) * ldk + kb * 32 + off;
  union { v16bf v; uint4 q[2]; } u;
  u.q[0] = *(const uint4*)(p);
  u.q[1] = *(const uint4*)(p + 16);
  return u.v;
}

DEV void store4bf(__bf16* dstp, float4 v) {
  bf4 t;
  t.x = (__bf16)v.x; t.y = (__bf16)v.y; t.z = (__bf16)v.z; t.w = (__bf16)v.w;
  *(bf4*)dstp = t;
}

#define WMMA_BF16(A, B, C) \
  __builtin_amdgcn_wmma_f32_16x16x32_bf16(false, (A), false, (B), (short)0, (C), false, false)

__global__ void __launch_bounds__(256, 1)
m3gnet_fused(const float* __restrict__ node_feat, const float* __restrict__ edge_feat,
             const float* __restrict__ rbf, const int* __restrict__ srcIdx,
             const int* __restrict__ dstIdx,
             const float* __restrict__ w_el1, const float* __restrict__ b_el1,
             const float* __restrict__ w_el2, const float* __restrict__ b_el2,
             const float* __restrict__ w_eg1, const float* __restrict__ b_eg1,
             const float* __restrict__ w_eg2, const float* __restrict__ b_eg2,
             const float* __restrict__ w_nl1, const float* __restrict__ b_nl1,
             const float* __restrict__ w_nl2, const float* __restrict__ b_nl2,
             const float* __restrict__ w_ng1, const float* __restrict__ b_ng1,
             const float* __restrict__ w_ng2, const float* __restrict__ b_ng2,
             const float* __restrict__ w_rbf_e, const float* __restrict__ w_rbf_n,
             float* __restrict__ e_out, float* __restrict__ v_out)
{
  extern __shared__ __align__(16) char smem[];
  __bf16* W_EL1 = (__bf16*)(smem + OFF_WEL1);
  __bf16* W_EG1 = (__bf16*)(smem + OFF_WEG1);
  __bf16* W_EL2 = (__bf16*)(smem + OFF_WEL2);
  __bf16* W_EG2 = (__bf16*)(smem + OFF_WEG2);
  __bf16* W_NL1 = (__bf16*)(smem + OFF_WNL1);
  __bf16* W_NG1 = (__bf16*)(smem + OFF_WNG1);
  __bf16* W_NL2 = (__bf16*)(smem + OFF_WNL2);
  __bf16* W_NG2 = (__bf16*)(smem + OFF_WNG2);
  float*  BIAS  = (float*)(smem + OFF_BIAS);
  float*  RBFW  = (float*)(smem + OFF_RBFW);   // [0..767] edge, [768..] node (stride 12)

  char*   pw    = smem + OFF_PW + (threadIdx.x >> 5) * PW_STRIDE;
  __bf16* Xw    = (__bf16*)(pw + PW_X);
  __bf16* Hw    = (__bf16*)(pw + PW_H);
  __bf16* Gw    = (__bf16*)(pw + PW_G);
  float*  RBFw  = (float*)(pw + PW_RBF);
  int*    DSTw  = (int*)(pw + PW_DST);

  // ------------- stage weights (block-cooperative, f32 -> bf16) -------------
  #define STAGE_W(dst_, src_, n_) \
    for (int i = threadIdx.x; i < (n_); i += blockDim.x) (dst_)[i] = (__bf16)(src_)[i];
  STAGE_W(W_EL1, w_el1, 12288) STAGE_W(W_EG1, w_eg1, 12288)
  STAGE_W(W_EL2, w_el2, 4096)  STAGE_W(W_EG2, w_eg2, 4096)
  STAGE_W(W_NL1, w_nl1, 12288) STAGE_W(W_NG1, w_ng1, 12288)
  STAGE_W(W_NL2, w_nl2, 4096)  STAGE_W(W_NG2, w_ng2, 4096)
  #undef STAGE_W
  #define STAGE_B(slot_, src_) \
    for (int i = threadIdx.x; i < 64; i += blockDim.x) BIAS[(slot_) * 64 + i] = (src_)[i];
  STAGE_B(B_EL1, b_el1) STAGE_B(B_EG1, b_eg1) STAGE_B(B_EL2, b_el2) STAGE_B(B_EG2, b_eg2)
  STAGE_B(B_NL1, b_nl1) STAGE_B(B_NG1, b_ng1) STAGE_B(B_NL2, b_nl2) STAGE_B(B_NG2, b_ng2)
  #undef STAGE_B
  for (int i = threadIdx.x; i < 64 * NRBF; i += blockDim.x) {
    int r = i / NRBF, k = i % NRBF;
    RBFW[r * 12 + k]       = w_rbf_e[i];
    RBFW[768 + r * 12 + k] = w_rbf_n[i];
  }

  // ------------- per-wave: gather X = [vi | vj | e] as bf16 -------------
  const int lane = threadIdx.x & 31;
  const long long base_e = (long long)blockIdx.x * EPB + (long long)(threadIdx.x >> 5) * 16;
  {
    int m = lane & 15;
    int p = (lane >> 4) & 1;                 // each half-wave does 32 of 64 features
    long long e = base_e + m;
    int si = srcIdx[e], di = dstIdx[e];
    const float* vi = node_feat + (long long)si * D + p * 32;
    const float* vj = node_feat + (long long)di * D + p * 32;
    const float* ef = edge_feat + e * D + p * 32;
    __bf16* xr = Xw + m * DIN;
    #pragma unroll
    for (int i = 0; i < 8; i++) {
      store4bf(xr + p * 32 + i * 4,           ((const float4*)vi)[i]);
      store4bf(xr + D + p * 32 + i * 4,       ((const float4*)vj)[i]);
      store4bf(xr + 2 * D + p * 32 + i * 4,   ((const float4*)ef)[i]);
    }
    if (lane < 16) {
      #pragma unroll
      for (int k = 0; k < NRBF; k++) RBFw[m * 12 + k] = rbf[e * NRBF + k];
      DSTw[m] = di;
    }
  }
  __syncthreads();

  const int ln  = threadIdx.x & 15;
  const int hi8 = (lane < 16) ? 0 : 8;

  // ==================================================================
  // phase = 0 : edge MLP (writes e_new to global and back into X)
  // phase = 1 : node MLP (scatter-add messages into v_out)
  // ==================================================================
  for (int phase = 0; phase < 2; phase++) {
    const __bf16* Wl1 = phase ? W_NL1 : W_EL1;
    const __bf16* Wg1 = phase ? W_NG1 : W_EG1;
    const __bf16* Wl2 = phase ? W_NL2 : W_EL2;
    const __bf16* Wg2 = phase ? W_NG2 : W_EG2;
    const int bl1 = phase ? B_NL1 : B_EL1, bg1 = phase ? B_NG1 : B_EG1;
    const int bl2 = phase ? B_NL2 : B_EL2, bg2 = phase ? B_NG2 : B_EG2;
    const int rbw = phase ? 768 : 0;

    // A fragments of X (16 x 192), 6 K-blocks of 32, kept in VGPRs
    v16bf A6[6];
    #pragma unroll
    for (int kb = 0; kb < 6; kb++) A6[kb] = lds_frag_a(Xw, DIN, kb);

    // ---- layer 1 (192 -> 64) : layers + gates branches, SiLU both ----
    for (int n = 0; n < 4; n++) {
      int d = n * 16 + ln;
      v8f ah, ag;
      float bh = BIAS[bl1 * 64 + d], bg = BIAS[bg1 * 64 + d];
      #pragma unroll
      for (int i = 0; i < 8; i++) { ah[i] = bh; ag[i] = bg; }
      #pragma unroll
      for (int kb = 0; kb < 6; kb++) {
        ah = WMMA_BF16(A6[kb], lds_frag_b(Wl1, DIN, n, kb), ah);
        ag = WMMA_BF16(A6[kb], lds_frag_b(Wg1, DIN, n, kb), ag);
      }
      #pragma unroll
      for (int i = 0; i < 8; i++) {
        Hw[(hi8 + i) * D + d] = (__bf16)silu_f(ah[i]);
        Gw[(hi8 + i) * D + d] = (__bf16)silu_f(ag[i]);
      }
    }
    __syncthreads();

    // ---- layer 2 (64 -> 64), SiLU(h) * sigmoid(g) * (rbf @ Wrbf^T) ----
    for (int n = 0; n < 4; n++) {
      int d = n * 16 + ln;
      v8f ah, ag;
      float bh = BIAS[bl2 * 64 + d], bg = BIAS[bg2 * 64 + d];
      #pragma unroll
      for (int i = 0; i < 8; i++) { ah[i] = bh; ag[i] = bg; }
      #pragma unroll
      for (int kb = 0; kb < 2; kb++) {
        ah = WMMA_BF16(lds_frag_a(Hw, D, kb), lds_frag_b(Wl2, D, n, kb), ah);
        ag = WMMA_BF16(lds_frag_a(Gw, D, kb), lds_frag_b(Wg2, D, n, kb), ag);
      }
      float rw[NRBF];
      #pragma unroll
      for (int k = 0; k < NRBF; k++) rw[k] = RBFW[rbw + d * 12 + k];
      #pragma unroll
      for (int i = 0; i < 8; i++) {
        int mm = hi8 + i;
        float r = 0.0f;
        #pragma unroll
        for (int k = 0; k < NRBF; k++) r = fmaf(RBFw[mm * 12 + k], rw[k], r);
        float val = silu_f(ah[i]) * sigm_f(ag[i]) * r;
        long long ee = base_e + mm;
        if (phase == 0) {
          float en = edge_feat[ee * D + d] + val;       // residual
          e_out[ee * D + d] = en;                       // output 1: e_new
          Xw[mm * DIN + 2 * D + d] = (__bf16)en;        // feed node MLP
        } else {
          unsafeAtomicAdd(&v_out[(long long)DSTw[mm] * D + d], val);  // scatter-sum
        }
      }
    }
    __syncthreads();
  }
}

__global__ void vnew_init(const float* __restrict__ nf, float* __restrict__ v_out) {
  int i = blockIdx.x * blockDim.x + threadIdx.x;
  if (i < NN * D) v_out[i] = nf[i];
}

extern "C" void kernel_launch(void* const* d_in, const int* in_sizes, int n_in,
                              void* d_out, int out_size, void* d_ws, size_t ws_size,
                              hipStream_t stream) {
  (void)in_sizes; (void)n_in; (void)out_size; (void)d_ws; (void)ws_size;
  const float* node_feat = (const float*)d_in[0];
  const float* edge_feat = (const float*)d_in[1];
  const float* rbf       = (const float*)d_in[2];
  const int*   src       = (const int*)d_in[3];
  const int*   dst       = (const int*)d_in[4];
  const float* e_l1_w = (const float*)d_in[5];  const float* e_l1_b = (const float*)d_in[6];
  const float* e_l2_w = (const float*)d_in[7];  const float* e_l2_b = (const float*)d_in[8];
  const float* e_g1_w = (const float*)d_in[9];  const float* e_g1_b = (const float*)d_in[10];
  const float* e_g2_w = (const float*)d_in[11]; const float* e_g2_b = (const float*)d_in[12];
  const float* n_l1_w = (const float*)d_in[13]; const float* n_l1_b = (const float*)d_in[14];
  const float* n_l2_w = (const float*)d_in[15]; const float* n_l2_b = (const float*)d_in[16];
  const float* n_g1_w = (const float*)d_in[17]; const float* n_g1_b = (const float*)d_in[18];
  const float* n_g2_w = (const float*)d_in[19]; const float* n_g2_b = (const float*)d_in[20];
  const float* edge_w_w = (const float*)d_in[21];
  const float* node_w_w = (const float*)d_in[22];

  float* e_out = (float*)d_out;                       // [E, D]
  float* v_out = e_out + E * (long long)D;            // [N, D]

  // v_new starts as node_feat; edge kernel atomically accumulates messages.
  vnew_init<<<(NN * D + 255) / 256, 256, 0, stream>>>(node_feat, v_out);

  static_cast<void>(hipFuncSetAttribute((const void*)m3gnet_fused,
                                        hipFuncAttributeMaxDynamicSharedMemorySize,
                                        SMEM_BYTES));
  m3gnet_fused<<<(int)(E / EPB), 256, SMEM_BYTES, stream>>>(
      node_feat, edge_feat, rbf, src, dst,
      e_l1_w, e_l1_b, e_l2_w, e_l2_b, e_g1_w, e_g1_b, e_g2_w, e_g2_b,
      n_l1_w, n_l1_b, n_l2_w, n_l2_b, n_g1_w, n_g1_b, n_g2_w, n_g2_b,
      edge_w_w, node_w_w, e_out, v_out);
}
